// GraphAttentionLayer_53661321396258
// MI455X (gfx1250) — matrix operand order repaired and checked by reading, not compiled, not run on hardware
//
#include <hip/hip_runtime.h>

typedef __attribute__((ext_vector_type(16))) __bf16 v16bf;
typedef __attribute__((ext_vector_type(8)))  float  v8f;
typedef __attribute__((ext_vector_type(4)))  float  v4f;

#define N_NODES 8192
#define F_IN    128
#define F_OUT   64
#define ALPHA   0.2f
#define LOG2E   1.44269504088896f

__device__ __forceinline__ __bf16 f2bf(float f) {
  unsigned u = __builtin_bit_cast(unsigned, f);
  unsigned r = u + 0x7fffu + ((u >> 16) & 1u);   // round-to-nearest-even
  unsigned short hb = (unsigned short)(r >> 16);
  return __builtin_bit_cast(__bf16, hb);
}

// ---------------------------------------------------------------------------
// Kernel 1: Wh = h@W  ->  bf16 Wh^T [64][8192] (K-contiguous for WMMA B),
//           Wh1 = log2(e) * Wh@a[:64], Wh2 = log2(e) * Wh@a[64:]
// (log2e pre-scaled so the main loop uses raw v_exp_f32; valid because
//  leakyrelu(c*x) = c*leakyrelu(x) for c > 0.)
// ---------------------------------------------------------------------------
__global__ void gat_precompute(const float* __restrict__ h,
                               const float* __restrict__ W,
                               const float* __restrict__ a,
                               unsigned short* __restrict__ WhT,
                               float* __restrict__ Wh1,
                               float* __restrict__ Wh2) {
  __shared__ float hrow[F_IN];
  __shared__ float red[F_OUT];
  const int i = blockIdx.x;
  const int c = threadIdx.x;                 // 0..63
  hrow[c]      = h[(size_t)i * F_IN + c];
  hrow[c + 64] = h[(size_t)i * F_IN + c + 64];
  __syncthreads();
  float acc = 0.f;
#pragma unroll
  for (int k = 0; k < F_IN; ++k)
    acc = __builtin_fmaf(hrow[k], W[k * F_OUT + c], acc);
  WhT[(size_t)c * N_NODES + i] = __builtin_bit_cast(unsigned short, f2bf(acc));

  red[c] = acc * a[c];
  __syncthreads();
  for (int s = 32; s > 0; s >>= 1) { if (c < s) red[c] += red[c + s]; __syncthreads(); }
  if (c == 0) Wh1[i] = red[0] * LOG2E;
  __syncthreads();
  red[c] = acc * a[F_OUT + c];
  __syncthreads();
  for (int s = 32; s > 0; s >>= 1) { if (c < s) red[c] += red[c + s]; __syncthreads(); }
  if (c == 0) Wh2[i] = red[0] * LOG2E;
}

// ---------------------------------------------------------------------------
// Kernel 2: fused  masked-softmax(leakyrelu(Wh1_i + Wh2_j))  @  Wh  -> ELU
// Flash-style: adj streamed ONCE (non-temporal), attention never materialized.
// One block = 16 output rows; 8 waves each own a 1024-wide j strip and carry
// four 16x16 f32 WMMA accumulators (N = 64) plus a 5th WMMA accumulator
// against an all-ones B matrix that produces the softmax row sums — this
// replaces a 16-long serial VALU add chain and sums the *same* bf16-rounded
// P values the numerator uses.
// ---------------------------------------------------------------------------
__global__ void __launch_bounds__(256)
gat_main(const float* __restrict__ adj,
         const unsigned short* __restrict__ WhT,
         const float* __restrict__ Wh1,
         const float* __restrict__ Wh2,
         float* __restrict__ out) {
  // phase 1: smem = Wh2 cache (8192 f32).  phase 2: smem = C partials (8192 f32) + row sums (128 f32)
  __shared__ __align__(16) float smem[N_NODES + 128];

  const int tid  = threadIdx.x;
  const int lane = tid & 31;
  const int wave = tid >> 5;
  const int half = lane >> 4;     // K-half per WMMA operand layout
  const int lm   = lane & 15;     // row (A) or column (B/C) within tile
  const int row0 = blockIdx.x * 16;

  for (int t = tid; t < N_NODES / 4; t += 256)
    ((v4f*)smem)[t] = ((const v4f*)Wh2)[t];
  __syncthreads();

  const float  wh1m   = Wh1[row0 + lm];
  const float* adjrow = adj + (size_t)(row0 + lm) * N_NODES;

  // all-ones bf16 B matrix for the row-sum WMMA
  v16bf Bones;
#pragma unroll
  for (int e = 0; e < 16; ++e)
    Bones[e] = __builtin_bit_cast(__bf16, (unsigned short)0x3F80);

  v8f acc0 = {}, acc1 = {}, acc2 = {}, acc3 = {}, accS = {};

  const int jbeg = wave * (N_NODES / 8);
  for (int js = 0; js < N_NODES / 8; js += 32) {
    const int j0 = jbeg + js;

    // ---- A tile: P[M][K] = adj * exp2(leakyrelu(Wh1_i + Wh2_j))  (bf16) ----
    // (Wh1/Wh2 pre-scaled by log2e; adj is exactly 0.0/1.0 so multiply == mask,
    //  keeping the loop branch-free — no per-element EXEC manipulation.)
    // A k order used here (contiguous, same as B): lane<16 holds K {0..15}; lane>=16 holds K {16..31}
    const v4f* ap = (const v4f*)(adjrow + j0);
    v4f q0 = __builtin_nontemporal_load(ap + half * 4 + 0);
    v4f q1 = __builtin_nontemporal_load(ap + half * 4 + 1);
    v4f q2 = __builtin_nontemporal_load(ap + half * 4 + 2);
    v4f q3 = __builtin_nontemporal_load(ap + half * 4 + 3);
    const v4f* wp = (const v4f*)(smem + j0);
    v4f w0 = wp[half * 4 + 0];
    v4f w1 = wp[half * 4 + 1];
    v4f w2 = wp[half * 4 + 2];
    v4f w3 = wp[half * 4 + 3];
    float adjv[16] = {q0.x,q0.y,q0.z,q0.w, q1.x,q1.y,q1.z,q1.w,
                      q2.x,q2.y,q2.z,q2.w, q3.x,q3.y,q3.z,q3.w};
    float w2v[16]  = {w0.x,w0.y,w0.z,w0.w, w1.x,w1.y,w1.z,w1.w,
                      w2.x,w2.y,w2.z,w2.w, w3.x,w3.y,w3.z,w3.w};
    v16bf A;
#pragma unroll
    for (int e = 0; e < 16; ++e) {
      float s  = wh1m + w2v[e];                        // log2-domain logit
      float ev = s > 0.f ? s : ALPHA * s;              // leaky relu (scale-invariant)
      float p  = __builtin_amdgcn_exp2f(ev) * adjv[e]; // branchless mask (adj in {0,1})
      A[e] = (__bf16)p;                                // native cvt (RNE)
    }

    // ---- B tiles: Wh[K][col] from bf16 Wh^T (K-contiguous) ----
    // B k order used here (contiguous, same as A): lanes 0-15 hold K=0..15, lanes 16-31 hold K=16..31, col = lane%16
    const size_t boff = (size_t)lm * N_NODES + (size_t)(j0 + half * 16);
    v16bf B0 = *(const v16bf*)(WhT + boff + (size_t) 0 * N_NODES);
    v16bf B1 = *(const v16bf*)(WhT + boff + (size_t)16 * N_NODES);
    v16bf B2 = *(const v16bf*)(WhT + boff + (size_t)32 * N_NODES);
    v16bf B3 = *(const v16bf*)(WhT + boff + (size_t)48 * N_NODES);

    acc0 = __builtin_amdgcn_wmma_f32_16x16x32_bf16(false, A, false, B0,    (short)0, acc0, false, false);
    acc1 = __builtin_amdgcn_wmma_f32_16x16x32_bf16(false, A, false, B1,    (short)0, acc1, false, false);
    acc2 = __builtin_amdgcn_wmma_f32_16x16x32_bf16(false, A, false, B2,    (short)0, acc2, false, false);
    acc3 = __builtin_amdgcn_wmma_f32_16x16x32_bf16(false, A, false, B3,    (short)0, acc3, false, false);
    accS = __builtin_amdgcn_wmma_f32_16x16x32_bf16(false, A, false, Bones, (short)0, accS, false, false);
  }

  __syncthreads();   // all waves done with the Wh2 region before reuse

  float* sC   = smem;              // [8 waves][16 rows][64 cols]
  float* sSum = smem + N_NODES;    // [8 waves][16 rows]
  {
    // C/D layout: VGPR r holds row M = r + 8*half, col = lane%16 (+16n per tile)
    float* base = sC + wave * 1024 + lm;
    const int mrow = 8 * half;
#pragma unroll
    for (int r = 0; r < 8; ++r) {
      base[(mrow + r) * 64 +  0] = acc0[r];
      base[(mrow + r) * 64 + 16] = acc1[r];
      base[(mrow + r) * 64 + 32] = acc2[r];
      base[(mrow + r) * 64 + 48] = acc3[r];
    }
    if (lm == 0) {                 // every column of accS holds the row sum
#pragma unroll
      for (int r = 0; r < 8; ++r)
        sSum[wave * 16 + mrow + r] = accS[r];
    }
  }
  __syncthreads();

  // combine 8 wave partials, normalize by row sum, ELU, store
#pragma unroll
  for (int k = 0; k < 4; ++k) {
    const int flat = tid + k * 256;        // 1024 outputs per block
    const int M    = flat >> 6;
    const int col  = flat & 63;
    float v = 0.f, l = 0.f;
#pragma unroll
    for (int w = 0; w < 8; ++w) {
      v += sC[w * 1024 + M * 64 + col];
      l += sSum[w * 16 + M];
    }
    const float hp = v / l;
    out[(size_t)(row0 + M) * F_OUT + col] =
        hp > 0.f ? hp : __builtin_amdgcn_exp2f(hp * LOG2E) - 1.f;
  }
}

// ---------------------------------------------------------------------------
extern "C" void kernel_launch(void* const* d_in, const int* in_sizes, int n_in,
                              void* d_out, int out_size, void* d_ws, size_t ws_size,
                              hipStream_t stream) {
  const float* h   = (const float*)d_in[0];
  const float* adj = (const float*)d_in[1];
  const float* W   = (const float*)d_in[2];
  const float* a   = (const float*)d_in[3];

  unsigned short* WhT = (unsigned short*)d_ws;                           // 64*8192 bf16 = 1 MB
  float* Wh1 = (float*)((char*)d_ws + (size_t)F_OUT * N_NODES * sizeof(unsigned short));
  float* Wh2 = Wh1 + N_NODES;

  gat_precompute<<<N_NODES, F_OUT, 0, stream>>>(h, W, a, WhT, Wh1, Wh2);
  gat_main<<<N_NODES / 16, 256, 0, stream>>>(adj, WhT, Wh1, Wh2, (float*)d_out);
}
